// GroupedLinear_14491219657137
// MI455X (gfx1250) — matrix operand order repaired and checked by reading, not compiled, not run on hardware
//
#include <hip/hip_runtime.h>

#define IN_F  512
#define OUT_F 512
#define NEXP  8
#define BM    128
#define KT    32
#define NTHREADS 512

typedef __attribute__((ext_vector_type(16))) __bf16          v16bf;
typedef __attribute__((ext_vector_type(8)))  __bf16          v8bf;
typedef __attribute__((ext_vector_type(8)))  float           v8f;
typedef __attribute__((ext_vector_type(8)))  unsigned short  u16x8;

union Frag { v16bf v; v8bf h[2]; };

__device__ __forceinline__ unsigned short f2bf(float f) {
  unsigned int u = __float_as_uint(f);
  u += 0x7FFFu + ((u >> 16) & 1u);        // round-to-nearest-even
  return (unsigned short)(u >> 16);
}
__device__ __forceinline__ float bf2f(unsigned short h) {
  return __uint_as_float(((unsigned int)h) << 16);
}

// ---------------------------------------------------------------------------
// Pass 1: split W (fp32) into hi/lo bf16 planes, K-tiled:
//   Wt[e][kb32][n][k']  with k' = k & 31, kb32 = k >> 5
// so the main kernel's B fragments are direct aligned b128 loads.
// ---------------------------------------------------------------------------
__global__ __launch_bounds__(256)
void split_w_kernel(const float* __restrict__ w,
                    unsigned short* __restrict__ wh,
                    unsigned short* __restrict__ wl) {
  const size_t i = (size_t)blockIdx.x * 256 + threadIdx.x;   // < 8*512*512
  const int k  = (int)(i & 511);
  const int n  = (int)((i >> 9) & 511);
  const int e  = (int)(i >> 18);
  const float f = w[i];
  const unsigned short h = f2bf(f);
  const unsigned short l = f2bf(f - bf2f(h));
  const size_t dst = ((((size_t)e * 16 + (k >> 5)) * 512 + n) << 5) + (k & 31);
  wh[dst] = h;
  wl[dst] = l;
}

// ---------------------------------------------------------------------------
// Pass 2: grouped GEMM, bf16x3 emulation of fp32.
// Block = 128 rows x 512 cols, 512 threads (16 waves, wave32).
// Wave (wm, wn): rows wm*32..+31 (2 row tiles), cols wn*128..+127 (8 col
// tiles) -> 16 accumulators; each B fragment feeds 2 row tiles (6 wmma per
// 4 b128 loads), halving L2 weight traffic vs 1 row tile.
// ---------------------------------------------------------------------------
__global__ __launch_bounds__(NTHREADS)
void grouped_gemm_bf16x3(const float* __restrict__ x,
                         const unsigned short* __restrict__ wh,
                         const unsigned short* __restrict__ wl,
                         const int*   __restrict__ tpe,
                         float*       __restrict__ out) {
  __shared__ alignas(16) unsigned short sAh[BM * KT];
  __shared__ alignas(16) unsigned short sAl[BM * KT];

  const int t    = threadIdx.x;
  const int lane = t & 31;        // wave32
  const int wid  = t >> 5;        // 16 waves
  const int wm   = wid & 3;       // 4 row groups of 32
  const int wn   = wid >> 2;      // 4 col groups of 128
  const int half = lane >> 4;
  const int mr   = lane & 15;

  const int r0 = blockIdx.x * BM;

  // uniform prefix sum of tokens_per_expert (scalar loads)
  int starts[NEXP + 1];
  int s = 0;
#pragma unroll
  for (int e = 0; e < NEXP; ++e) { starts[e] = s; s += tpe[e]; }
  starts[NEXP] = s;

  for (int e = 0; e < NEXP; ++e) {
    const int lo = starts[e]     > r0      ? starts[e]     : r0;
    const int hi = starts[e + 1] < r0 + BM ? starts[e + 1] : r0 + BM;
    if (lo >= hi) continue;                       // uniform per block

    v8f acc[2][8] = {};

    for (int kb32 = 0; kb32 < IN_F / KT; ++kb32) {
      // ---- stage A tile (128 x 32 fp32 -> hi/lo bf16), 8 elems/thread ----
      {
        const int row = t >> 2;          // 0..127
        const int kc  = (t & 3) * 8;     // 0,8,16,24
        const float* src = x + (size_t)(r0 + row) * IN_F + kb32 * KT + kc;
        const float4 f0 = *(const float4*)src;
        const float4 f1 = *(const float4*)(src + 4);
        const float f[8] = {f0.x, f0.y, f0.z, f0.w, f1.x, f1.y, f1.z, f1.w};
        u16x8 h8, l8;
#pragma unroll
        for (int i = 0; i < 8; ++i) {
          const unsigned short hh = f2bf(f[i]);
          h8[i] = hh;
          l8[i] = f2bf(f[i] - bf2f(hh));
        }
        *(u16x8*)&sAh[row * KT + kc] = h8;
        *(u16x8*)&sAl[row * KT + kc] = l8;
      }
      __syncthreads();

      // ---- A fragments (2 row tiles) per CDNA5 16-bit A(16x32) layout ----
      Frag aH[2], aL[2];
#pragma unroll
      for (int rt = 0; rt < 2; ++rt) {
        const int arow = wm * 32 + rt * 16 + mr;
        aH[rt].h[0] = *(const v8bf*)&sAh[arow * KT + half * 8];
        aH[rt].h[1] = *(const v8bf*)&sAh[arow * KT + 16 + half * 8];
        aL[rt].h[0] = *(const v8bf*)&sAl[arow * KT + half * 8];
        aL[rt].h[1] = *(const v8bf*)&sAl[arow * KT + 16 + half * 8];
      }

      // ---- B fragments straight from L2-resident tiled weights ----
      const size_t tbase = ((size_t)(e * 16 + kb32)) << 14;   // *512*32
      const unsigned short* bh = wh + tbase;
      const unsigned short* bl = wl + tbase;

#pragma unroll
      for (int tt = 0; tt < 8; ++tt) {
        const int n = wn * 128 + tt * 16 + mr;
        Frag bH, bL;
        bH.h[0] = *(const v8bf*)(bh + n * KT + half * 8);
        bH.h[1] = *(const v8bf*)(bh + n * KT + 16 + half * 8);
        bL.h[0] = *(const v8bf*)(bl + n * KT + half * 8);
        bL.h[1] = *(const v8bf*)(bl + n * KT + 16 + half * 8);
#pragma unroll
        for (int rt = 0; rt < 2; ++rt) {
          acc[rt][tt] = __builtin_amdgcn_wmma_f32_16x16x32_bf16(
              false, aL[rt].v, false, bH.v, (short)0, acc[rt][tt], false, false);
          acc[rt][tt] = __builtin_amdgcn_wmma_f32_16x16x32_bf16(
              false, aH[rt].v, false, bL.v, (short)0, acc[rt][tt], false, false);
          acc[rt][tt] = __builtin_amdgcn_wmma_f32_16x16x32_bf16(
              false, aH[rt].v, false, bH.v, (short)0, acc[rt][tt], false, false);
        }
      }
      __syncthreads();
    }

    // ---- store ----
    if (lo == r0 && hi == r0 + BM) {
      // fast path: entire 128-row panel belongs to expert e (common case)
#pragma unroll
      for (int rt = 0; rt < 2; ++rt) {
#pragma unroll
        for (int tt = 0; tt < 8; ++tt) {
          const int col = wn * 128 + tt * 16 + mr;
#pragma unroll
          for (int i = 0; i < 8; ++i) {
            const int grow = r0 + wm * 32 + rt * 16 + half * 8 + i;
            out[(size_t)grow * OUT_F + col] = acc[rt][tt][i];
          }
        }
      }
    } else {
      // rare: panel straddles an expert boundary -> predicate by row
#pragma unroll
      for (int rt = 0; rt < 2; ++rt) {
#pragma unroll
        for (int tt = 0; tt < 8; ++tt) {
          const int col = wn * 128 + tt * 16 + mr;
#pragma unroll
          for (int i = 0; i < 8; ++i) {
            const int grow = r0 + wm * 32 + rt * 16 + half * 8 + i;
            if (grow >= lo && grow < hi)
              out[(size_t)grow * OUT_F + col] = acc[rt][tt][i];
          }
        }
      }
    }
  }
}

extern "C" void kernel_launch(void* const* d_in, const int* in_sizes, int n_in,
                              void* d_out, int out_size, void* d_ws, size_t ws_size,
                              hipStream_t stream) {
  (void)n_in; (void)out_size; (void)ws_size;
  const float* x   = (const float*)d_in[0];
  const float* w   = (const float*)d_in[1];
  const int*   tpe = (const int*)d_in[2];
  float*       out = (float*)d_out;

  const size_t wElems = (size_t)NEXP * OUT_F * IN_F;          // 2M
  unsigned short* wh = (unsigned short*)d_ws;
  unsigned short* wl = wh + wElems;                           // 8 MB total

  // Pass 1: split + tile weights (cheap; L2-resident afterwards)
  split_w_kernel<<<dim3((unsigned)(wElems / 256)), dim3(256), 0, stream>>>(w, wh, wl);

  // Pass 2: grouped GEMM
  const int T = in_sizes[0] / IN_F;                           // 131072
  grouped_gemm_bf16x3<<<dim3(T / BM), dim3(NTHREADS), 0, stream>>>(x, wh, wl, tpe, out);
}
